// NodePairBlock_60138132078616
// MI455X (gfx1250) — matrix operand order repaired and checked by reading, not compiled, not run on hardware
//
#include <hip/hip_runtime.h>
#include <math.h>

typedef __attribute__((ext_vector_type(16))) _Float16 v16h;
typedef __attribute__((ext_vector_type(8)))  float    v8f;

#define BB 4
#define NN 256
#define DD 128

// ---- workspace layout (bytes) ----
#define WS_PFNW1T   0                                  // [256][640] f16
#define WS_PFNW2T   (WS_PFNW1T + 256*640*2)            // [128][256] f16
#define WS_PFFW1T   (WS_PFNW2T + 128*256*2)            // [512][128] f16
#define WS_PFFW2T   (WS_PFFW1T + 512*128*2)            // [128][512] f16
#define WS_NODE16   (WS_PFFW2T + 128*512*2)            // [B*N][128] f16
#define WS_OCTX     (WS_NODE16 + BB*NN*DD*2)           // [B*N][128] f32
#define WS_ICTX     (WS_OCTX   + BB*NN*DD*4)           // [B*N][128] f32

__device__ __forceinline__ float gelu_exact(float x) {
  return 0.5f * x * (1.0f + erff(x * 0.70710678118654752f));
}
__device__ __forceinline__ float sigmoid_f(float x) {
  return 1.0f / (1.0f + expf(-x));
}
__device__ __forceinline__ v8f zero8() {
  v8f z;
#pragma unroll
  for (int i = 0; i < 8; ++i) z[i] = 0.f;
  return z;
}

// A fragment (16x32 f16, M x K): lane<16 -> row=lane, K {0..7,16..23};
// lane>=16 -> same row, K {8..15,24..31}
__device__ __forceinline__ v16h frag_a_lds(const _Float16* base, int ld, int lane) {
  const int r  = lane & 15;
  const int kb = (lane >> 4) << 3;
  const _Float16* p = base + r * ld + kb;
  v16h f;
#pragma unroll
  for (int t = 0; t < 8; ++t) { f[t] = p[t]; f[t + 8] = p[t + 16]; }
  return f;
}
__device__ __forceinline__ v16h frag_a_lds_f32(const float* base, int ld, int lane) {
  const int r  = lane & 15;
  const int kb = (lane >> 4) << 3;
  const float* p = base + r * ld + kb;
  v16h f;
#pragma unroll
  for (int t = 0; t < 8; ++t) { f[t] = (_Float16)p[t]; f[t + 8] = (_Float16)p[t + 16]; }
  return f;
}
// B fragment (32x16 f16, K x N) from B^T layout [n][k] (K contiguous):
// lane<16 -> col=lane, K {0..7,16..23} ; lane>=16 -> col=lane-16, K {8..15,24..31}
__device__ __forceinline__ v16h frag_b_g(const _Float16* base, int ld, int lane) {
  const _Float16* p = base + (size_t)(lane & 15) * ld + ((lane >> 4) << 3);
  v16h f;
#pragma unroll
  for (int t = 0; t < 8; ++t) { f[t] = p[t]; f[t + 8] = p[t + 16]; }
  return f;
}
__device__ __forceinline__ v8f wmma_f16(v16h a, v16h b, v8f c) {
  return __builtin_amdgcn_wmma_f32_16x16x32_f16(false, a, false, b, (short)0, c, false, false);
}

// ---------------- weight prep: transpose f32 [K][N] -> f16 [N][K] ----------------
__global__ void k_transpose_f16(_Float16* __restrict__ dst, const float* __restrict__ src,
                                int K, int Ncols) {
  int idx = blockIdx.x * blockDim.x + threadIdx.x;
  if (idx >= K * Ncols) return;
  int n = idx / K;
  int k = idx - n * K;
  dst[(size_t)n * K + k] = (_Float16)src[(size_t)k * Ncols + n];
}

// ---------------- context reductions ----------------
__global__ void k_out_ctx(float* __restrict__ octx, const float* __restrict__ pair,
                          const int* __restrict__ em) {
  const int bi = blockIdx.x;        // b*N + i
  const int d  = threadIdx.x;
  const float* pr = pair + (size_t)bi * NN * DD;
  const int*   mr = em   + (size_t)bi * NN;
  float s = 0.f, deg = 0.f;
  for (int j = 0; j < NN; ++j) {
    const float m = (float)mr[j];
    deg += m;
    s   += m * pr[(size_t)j * DD + d];
  }
  octx[(size_t)bi * DD + d] = s / fmaxf(deg, 1.0f);
}

__global__ void k_in_ctx(float* __restrict__ ictx, const float* __restrict__ pair,
                         const int* __restrict__ em) {
  const int b = blockIdx.x >> 8;
  const int j = blockIdx.x & 255;
  const int d = threadIdx.x;
  const float* pb = pair + (size_t)b * NN * NN * DD;
  const int*   mb = em   + (size_t)b * NN * NN;
  float s = 0.f, deg = 0.f;
  for (int i = 0; i < NN; ++i) {
    const float m = (float)mb[(size_t)i * NN + j];
    deg += m;
    s   += m * pb[((size_t)i * NN + j) * DD + d];
  }
  ictx[((size_t)b * NN + j) * DD + d] = s / fmaxf(deg, 1.0f);
}

// ---------------- fused node path (one block per (b,i) row) ----------------
__global__ __launch_bounds__(DD) void k_node(
    float* __restrict__ out_node, _Float16* __restrict__ node16,
    const float* __restrict__ node_h, const float* __restrict__ octx,
    const float* __restrict__ ictx, const int* __restrict__ node_mask,
    const float* __restrict__ ogW, const float* __restrict__ ogb,
    const float* __restrict__ igW, const float* __restrict__ igb,
    const float* __restrict__ W1, const float* __restrict__ b1,
    const float* __restrict__ W2, const float* __restrict__ b2,
    const float* __restrict__ fW1, const float* __restrict__ fb1,
    const float* __restrict__ fW2, const float* __restrict__ fb2,
    const float* __restrict__ g1, const float* __restrict__ be1,
    const float* __restrict__ g2, const float* __restrict__ be2) {
  __shared__ float sx[DD], so[DD], si[DD], su[3 * DD], sh1[2 * DD], sy[DD], sh2[4 * DD], st[DD];
  __shared__ float stats[2];
  const int bi = blockIdx.x;
  const int d  = threadIdx.x;
  const size_t row = (size_t)bi * DD;

  sx[d] = node_h[row + d];
  so[d] = octx[row + d];
  si[d] = ictx[row + d];
  __syncthreads();

  // gates
  float ao = ogb[d], ai = igb[d];
  for (int k = 0; k < DD; ++k) {
    ao += sx[k] * ogW[k * DD + d];
    ai += sx[k] * igW[k * DD + d];
  }
  for (int k = 0; k < DD; ++k) {
    ao += so[k] * ogW[(DD + k) * DD + d];
    ai += si[k] * igW[(DD + k) * DD + d];
  }
  su[d]          = sx[d];
  su[DD + d]     = sigmoid_f(ao) * so[d];
  su[2 * DD + d] = sigmoid_f(ai) * si[d];
  __syncthreads();

  // nfp: 384 -> 256 (gelu) -> 128
  for (int r = 0; r < 2; ++r) {
    const int e = d + r * DD;
    float a = b1[e];
    for (int k = 0; k < 3 * DD; ++k) a += su[k] * W1[k * (2 * DD) + e];
    sh1[e] = gelu_exact(a);
  }
  __syncthreads();
  float u = b2[d];
  for (int k = 0; k < 2 * DD; ++k) u += sh1[k] * W2[k * DD + d];
  const float t1 = sx[d] + u;

  st[d] = t1;
  __syncthreads();
  if (d == 0) {
    float m = 0.f;
    for (int k = 0; k < DD; ++k) m += st[k];
    m *= (1.0f / DD);
    float vv = 0.f;
    for (int k = 0; k < DD; ++k) { const float dd2 = st[k] - m; vv += dd2 * dd2; }
    vv *= (1.0f / DD);
    stats[0] = m;
    stats[1] = rsqrtf(vv + 1e-5f);
  }
  __syncthreads();
  const float y = (t1 - stats[0]) * stats[1] * g1[d] + be1[d];
  sy[d] = y;
  __syncthreads();

  // nff: 128 -> 512 (gelu) -> 128
  for (int r = 0; r < 4; ++r) {
    const int e = d + r * DD;
    float a = fb1[e];
    for (int k = 0; k < DD; ++k) a += sy[k] * fW1[k * (4 * DD) + e];
    sh2[e] = gelu_exact(a);
  }
  __syncthreads();
  float u2 = fb2[d];
  for (int k = 0; k < 4 * DD; ++k) u2 += sh2[k] * fW2[k * DD + d];
  const float t2 = y + u2;

  st[d] = t2;
  __syncthreads();
  if (d == 0) {
    float m = 0.f;
    for (int k = 0; k < DD; ++k) m += st[k];
    m *= (1.0f / DD);
    float vv = 0.f;
    for (int k = 0; k < DD; ++k) { const float dd2 = st[k] - m; vv += dd2 * dd2; }
    vv *= (1.0f / DD);
    stats[0] = m;
    stats[1] = rsqrtf(vv + 1e-5f);
  }
  __syncthreads();
  float z = (t2 - stats[0]) * stats[1] * g2[d] + be2[d];
  z *= (float)node_mask[bi];
  out_node[row + d] = z;
  node16[row + d]   = (_Float16)z;
}

// ---------------- fused pair path: one block = 64 j-rows of a (b,i) slab ----------------
__global__ __launch_bounds__(128) void k_pair(
    float* __restrict__ outPair, const float* __restrict__ pair_h,
    const int* __restrict__ edge_mask, const _Float16* __restrict__ node16,
    const _Float16* __restrict__ W1t,  // pfn_W1^T f16 [256][640]
    const _Float16* __restrict__ W2t,  // pfn_W2^T f16 [128][256]
    const _Float16* __restrict__ F1t,  // pff_W1^T f16 [512][128]
    const _Float16* __restrict__ F2t,  // pff_W2^T f16 [128][512]
    const float* __restrict__ b1, const float* __restrict__ b2,
    const float* __restrict__ fb1, const float* __restrict__ fb2,
    const float* __restrict__ g1, const float* __restrict__ be1,
    const float* __restrict__ g2, const float* __restrict__ be2) {
  __shared__ _Float16 featA[64 * 34];   // K-chunk of feat, stride 34 (conflict-free frags)
  __shared__ _Float16 Hst[64 * 130];    // gelu hidden staging (wave-private rows), stride 130
  __shared__ float    P1[64 * 129];     // fp32 row buffer for LN, stride 129

  const int t    = threadIdx.x;
  const int lane = t & 31;
  const int wave = t >> 5;           // 4 waves
  const int mrow0 = wave * 16;       // each wave owns one 16-row m-tile
  const int cl   = lane & 15;        // column-within-tile for C/D layout
  const int rsel = lane >> 4;        // 0/1: row offset 0/8 for C/D layout

  const int blk = blockIdx.x;
  const int q   = blk & 3;           // row quarter
  const int bi  = blk >> 2;          // b*N + i
  const int b   = bi >> 8;
  const int i   = bi & 255;
  const int j0  = q * 64;

  const float*    pr = pair_h + (((size_t)bi) * NN + j0) * DD;   // [64][128]
  const _Float16* nb = node16 + (size_t)b * NN * DD;
  const _Float16* hi = nb + (size_t)i * DD;

  // ===== pfn MLP: hidden(256) = gelu(feat(640) @ W1 + b1); upd(128) = hidden @ W2 =====
  v8f upd[8];
#pragma unroll
  for (int n = 0; n < 8; ++n) upd[n] = zero8();

  for (int hf = 0; hf < 2; ++hf) {     // hidden half: 128 cols
    v8f hid[8];
#pragma unroll
    for (int n = 0; n < 8; ++n) hid[n] = zero8();

    for (int kc = 0; kc < 20; ++kc) {  // K = 640 in chunks of 32
      __syncthreads();
      const int kBase = kc * 32;
      for (int e = t; e < 64 * 32; e += 128) {
        const int r = e >> 5, c = e & 31, k = kBase + c;
        const int j = j0 + r;
        float val;
        if (k < 128)      val = pr[(size_t)r * DD + k];
        else if (k < 256) val = (float)hi[k - 128];
        else if (k < 384) val = (float)nb[(size_t)j * DD + (k - 256)];
        else if (k < 512) val = (float)hi[k - 384] - (float)nb[(size_t)j * DD + (k - 384)];
        else              val = (float)hi[k - 512] * (float)nb[(size_t)j * DD + (k - 512)];
        featA[r * 34 + c] = (_Float16)val;
      }
      __syncthreads();
      const v16h a = frag_a_lds(featA + mrow0 * 34, 34, lane);
#pragma unroll
      for (int nt = 0; nt < 8; ++nt) {
        const v16h bf = frag_b_g(W1t + (size_t)(hf * 128 + nt * 16) * 640 + kBase, 640, lane);
        hid[nt] = wmma_f16(a, bf, hid[nt]);
      }
    }
    // bias + gelu -> f16 staging (wave-private rows)
#pragma unroll
    for (int nt = 0; nt < 8; ++nt) {
      const float bias = b1[hf * 128 + nt * 16 + cl];
#pragma unroll
      for (int vr = 0; vr < 8; ++vr) {
        const int r = mrow0 + vr + 8 * rsel;
        Hst[r * 130 + nt * 16 + cl] = (_Float16)gelu_exact(hid[nt][vr] + bias);
      }
    }
    // GEMM2: upd += H @ W2[hf*128 : hf*128+128]
    for (int kc = 0; kc < 4; ++kc) {
      const v16h a = frag_a_lds(Hst + mrow0 * 130 + kc * 32, 130, lane);
#pragma unroll
      for (int nt = 0; nt < 8; ++nt) {
        const v16h bf = frag_b_g(W2t + (size_t)(nt * 16) * 256 + hf * 128 + kc * 32, 256, lane);
        upd[nt] = wmma_f16(a, bf, upd[nt]);
      }
    }
  }

  // residual + bias -> P1 (fp32)
  __syncthreads();
#pragma unroll
  for (int nt = 0; nt < 8; ++nt) {
    const int c = nt * 16 + cl;
    const float bias = b2[c];
#pragma unroll
    for (int vr = 0; vr < 8; ++vr) {
      const int r = mrow0 + vr + 8 * rsel;
      P1[r * 129 + c] = upd[nt][vr] + bias + pr[(size_t)r * DD + c];
    }
  }
  __syncthreads();
  // LayerNorm 1 (one thread per row; stride-129 keeps banks rotating)
  if (t < 64) {
    float* row = P1 + t * 129;
    float m = 0.f;
    for (int c = 0; c < DD; ++c) m += row[c];
    m *= (1.0f / DD);
    float vv = 0.f;
    for (int c = 0; c < DD; ++c) { const float dd2 = row[c] - m; vv += dd2 * dd2; }
    vv *= (1.0f / DD);
    const float inv = rsqrtf(vv + 1e-5f);
    for (int c = 0; c < DD; ++c) row[c] = (row[c] - m) * inv * g1[c] + be1[c];
  }
  __syncthreads();

  // ===== pff MLP: h(512) = gelu(P1(128) @ F1 + fb1); u2(128) = h @ F2 =====
  v8f u2[8];
#pragma unroll
  for (int n = 0; n < 8; ++n) u2[n] = zero8();

  for (int grp = 0; grp < 4; ++grp) {   // hidden groups of 128
    v8f hh[8];
#pragma unroll
    for (int n = 0; n < 8; ++n) hh[n] = zero8();
    for (int kc = 0; kc < 4; ++kc) {
      const v16h a = frag_a_lds_f32(P1 + mrow0 * 129 + kc * 32, 129, lane);
#pragma unroll
      for (int nt = 0; nt < 8; ++nt) {
        const v16h bf = frag_b_g(F1t + (size_t)(grp * 128 + nt * 16) * 128 + kc * 32, 128, lane);
        hh[nt] = wmma_f16(a, bf, hh[nt]);
      }
    }
#pragma unroll
    for (int nt = 0; nt < 8; ++nt) {
      const float bias = fb1[grp * 128 + nt * 16 + cl];
#pragma unroll
      for (int vr = 0; vr < 8; ++vr) {
        const int r = mrow0 + vr + 8 * rsel;
        Hst[r * 130 + nt * 16 + cl] = (_Float16)gelu_exact(hh[nt][vr] + bias);
      }
    }
    for (int kc = 0; kc < 4; ++kc) {
      const v16h a = frag_a_lds(Hst + mrow0 * 130 + kc * 32, 130, lane);
#pragma unroll
      for (int nt = 0; nt < 8; ++nt) {
        const v16h bf = frag_b_g(F2t + (size_t)(nt * 16) * 512 + grp * 128 + kc * 32, 512, lane);
        u2[nt] = wmma_f16(a, bf, u2[nt]);
      }
    }
  }

  // t2 = P1 + u2 + fb2  (read residual first, then overwrite P1)
#pragma unroll
  for (int nt = 0; nt < 8; ++nt) {
    const int c = nt * 16 + cl;
    const float bias = fb2[c];
#pragma unroll
    for (int vr = 0; vr < 8; ++vr) {
      const int r = mrow0 + vr + 8 * rsel;
      u2[nt][vr] += bias + P1[r * 129 + c];
    }
  }
  __syncthreads();
#pragma unroll
  for (int nt = 0; nt < 8; ++nt) {
    const int c = nt * 16 + cl;
#pragma unroll
    for (int vr = 0; vr < 8; ++vr) {
      const int r = mrow0 + vr + 8 * rsel;
      P1[r * 129 + c] = u2[nt][vr];
    }
  }
  __syncthreads();
  // LayerNorm 2 + edge mask
  if (t < 64) {
    float* row = P1 + t * 129;
    float m = 0.f;
    for (int c = 0; c < DD; ++c) m += row[c];
    m *= (1.0f / DD);
    float vv = 0.f;
    for (int c = 0; c < DD; ++c) { const float dd2 = row[c] - m; vv += dd2 * dd2; }
    vv *= (1.0f / DD);
    const float inv = rsqrtf(vv + 1e-5f);
    const float msk = (float)edge_mask[(size_t)bi * NN + j0 + t];
    for (int c = 0; c < DD; ++c) row[c] = ((row[c] - m) * inv * g2[c] + be2[c]) * msk;
  }
  __syncthreads();
  // coalesced store
  float* outp = outPair + (((size_t)bi) * NN + j0) * DD;
  for (int e = t; e < 64 * DD; e += 128) {
    const int r = e >> 7, c = e & 127;
    outp[(size_t)r * DD + c] = P1[r * 129 + c];
  }
}

extern "C" void kernel_launch(void* const* d_in, const int* in_sizes, int n_in,
                              void* d_out, int out_size, void* d_ws, size_t ws_size,
                              hipStream_t stream) {
  const float* node_h    = (const float*)d_in[0];
  const float* pair_h    = (const float*)d_in[1];
  const int*   node_mask = (const int*)d_in[2];
  const int*   edge_mask = (const int*)d_in[3];
  const float* nfp_W1 = (const float*)d_in[4];
  const float* nfp_b1 = (const float*)d_in[5];
  const float* nfp_W2 = (const float*)d_in[6];
  const float* nfp_b2 = (const float*)d_in[7];
  const float* pfn_W1 = (const float*)d_in[8];
  const float* pfn_b1 = (const float*)d_in[9];
  const float* pfn_W2 = (const float*)d_in[10];
  const float* pfn_b2 = (const float*)d_in[11];
  const float* nff_W1 = (const float*)d_in[12];
  const float* nff_b1 = (const float*)d_in[13];
  const float* nff_W2 = (const float*)d_in[14];
  const float* nff_b2 = (const float*)d_in[15];
  const float* pff_W1 = (const float*)d_in[16];
  const float* pff_b1 = (const float*)d_in[17];
  const float* pff_W2 = (const float*)d_in[18];
  const float* pff_b2 = (const float*)d_in[19];
  const float* og_W = (const float*)d_in[20];
  const float* og_b = (const float*)d_in[21];
  const float* ig_W = (const float*)d_in[22];
  const float* ig_b = (const float*)d_in[23];
  const float* nn1_g = (const float*)d_in[24];
  const float* nn1_b = (const float*)d_in[25];
  const float* nn2_g = (const float*)d_in[26];
  const float* nn2_b = (const float*)d_in[27];
  const float* pn1_g = (const float*)d_in[28];
  const float* pn1_b = (const float*)d_in[29];
  const float* pn2_g = (const float*)d_in[30];
  const float* pn2_b = (const float*)d_in[31];

  char* ws = (char*)d_ws;
  _Float16* W1t    = (_Float16*)(ws + WS_PFNW1T);
  _Float16* W2t    = (_Float16*)(ws + WS_PFNW2T);
  _Float16* F1t    = (_Float16*)(ws + WS_PFFW1T);
  _Float16* F2t    = (_Float16*)(ws + WS_PFFW2T);
  _Float16* node16 = (_Float16*)(ws + WS_NODE16);
  float*    octx   = (float*)(ws + WS_OCTX);
  float*    ictx   = (float*)(ws + WS_ICTX);

  float* out_node = (float*)d_out;
  float* out_pair = (float*)d_out + (size_t)BB * NN * DD;

  // weight prep (f16, transposed to [N][K])
  k_transpose_f16<<<(640 * 256 + 255) / 256, 256, 0, stream>>>(W1t, pfn_W1, 640, 256);
  k_transpose_f16<<<(256 * 128 + 255) / 256, 256, 0, stream>>>(W2t, pfn_W2, 256, 128);
  k_transpose_f16<<<(128 * 512 + 255) / 256, 256, 0, stream>>>(F1t, pff_W1, 128, 512);
  k_transpose_f16<<<(512 * 128 + 255) / 256, 256, 0, stream>>>(F2t, pff_W2, 512, 128);

  // context reductions
  k_out_ctx<<<BB * NN, DD, 0, stream>>>(octx, pair_h, edge_mask);
  k_in_ctx<<<BB * NN, DD, 0, stream>>>(ictx, pair_h, edge_mask);

  // node path (writes node output + f16 copy for pair stage)
  k_node<<<BB * NN, DD, 0, stream>>>(out_node, node16, node_h, octx, ictx, node_mask,
                                     og_W, og_b, ig_W, ig_b,
                                     nfp_W1, nfp_b1, nfp_W2, nfp_b2,
                                     nff_W1, nff_b1, nff_W2, nff_b2,
                                     nn1_g, nn1_b, nn2_g, nn2_b);

  // pair path: 4 blocks per (b,i) slab (64 rows each)
  k_pair<<<BB * NN * 4, 128, 0, stream>>>(out_pair, pair_h, edge_mask, node16,
                                          W1t, W2t, F1t, F2t,
                                          pfn_b1, pfn_b2, pff_b1, pff_b2,
                                          pn1_g, pn1_b, pn2_g, pn2_b);
}
